// Attention_69604239999500
// MI455X (gfx1250) — matrix operand order repaired and checked by reading, not compiled, not run on hardware
//
#include <hip/hip_runtime.h>
#include <hip/hip_bf16.h>

// Problem constants (from reference): B=32, S=2048, D=1024, U=1024
#define BB 32
#define SS 2048
#define DD 1024
#define UU 1024
#define KT (DD / 32)   // 32 K-tiles of 32 along D
#define NT (UU / 16)   // 64 N-tiles of 16 along U

typedef __attribute__((ext_vector_type(16))) __bf16 v16bf;
typedef __attribute__((ext_vector_type(8)))  float  v8f;

__device__ __forceinline__ unsigned short f32_to_bf16(float f) {
    unsigned int u = __float_as_uint(f);
    // round-to-nearest-even
    u = (u + 0x7FFFu + ((u >> 16) & 1u)) >> 16;
    return (unsigned short)u;
}

// ---------------------------------------------------------------------------
// Kernel 0: convert + swizzle W1 [D,U] fp32 (row major, d-major) into WMMA
// B-fragment order, bf16:  W1p[((nt*KT + kt)*32 + lane)*16 + e]
// B-matrix 32x16 layout (wave32): lane -> N = lane%16, half h = lane/16,
// element e -> K = h*16 + e   (contiguous k order; A in k_score uses the same)
// ---------------------------------------------------------------------------
__global__ __launch_bounds__(256) void k_w1pack(const float* __restrict__ W1,
                                                unsigned short* __restrict__ W1p) {
    unsigned idx = blockIdx.x * 256u + threadIdx.x;      // 0 .. D*U-1
    unsigned e  = idx & 15u;
    unsigned L  = (idx >> 4) & 31u;
    unsigned kt = (idx >> 9) & 31u;
    unsigned nt = idx >> 14;                              // 0..63
    unsigned h  = L >> 4;
    unsigned n  = L & 15u;
    unsigned K  = h * 16u + e;
    unsigned d  = kt * 32u + K;
    unsigned u  = nt * 16u + n;
    W1p[idx] = f32_to_bf16(W1[(size_t)d * UU + u]);
}

// ---------------------------------------------------------------------------
// Kernel 1: g[b,u] = hid[b,:] . W2[:,u] + b1[u] + b2[u]   (tiny: 67 MFLOP)
// ---------------------------------------------------------------------------
__global__ __launch_bounds__(256) void k_g(const float* __restrict__ hid,
                                           const float* __restrict__ W2,
                                           const float* __restrict__ b1,
                                           const float* __restrict__ b2,
                                           float* __restrict__ g) {
    __shared__ float sH[DD];
    const int b = blockIdx.y;
    const int tid = threadIdx.x;
    for (int i = tid; i < DD; i += 256) sH[i] = hid[b * DD + i];
    __syncthreads();
    const int u = blockIdx.x * 256 + tid;
    float acc = 0.f;
    #pragma unroll 4
    for (int d = 0; d < DD; ++d) acc = fmaf(sH[d], W2[(size_t)d * UU + u], acc);
    g[b * UU + u] = acc + b1[u] + b2[u];
}

// ---------------------------------------------------------------------------
// Kernel 2 (main): per (b, 16-row s-tile) compute
//   score[b,s] = V . tanh(enc[b,s,:] @ W1 + g[b,:]) + bv
// 256 threads = 8 waves; wave w handles N-tiles w*8 .. w*8+7.
// Hot loop fully unrolled so all B-fragment loads use SADDR/immediate
// offsets (no per-iteration 64-bit VALU address math).
// ---------------------------------------------------------------------------
__global__ __launch_bounds__(256) void k_score(const float* __restrict__ enc,
                                               const unsigned short* __restrict__ W1p,
                                               const float* __restrict__ g,
                                               const float* __restrict__ V,
                                               const float* __restrict__ bvp,
                                               float* __restrict__ sc) {
    __shared__ float rawA[16 * DD];                           // 64 KB fp32 enc tile
    __shared__ __align__(32) unsigned short sA[KT * 32 * 16]; // 32 KB bf16 A-frags
    __shared__ float sG[UU];                                  // 4 KB
    __shared__ float sV[UU];                                  // 4 KB
    __shared__ float sScore[16];

    const int tid  = threadIdx.x;
    const int b    = blockIdx.y;
    const int s0   = blockIdx.x * 16;
    const int wave = tid >> 5;
    const int lane = tid & 31;

    // Phase 1: coalesced b128 load of the 16 x 1024 enc tile (fp32)
    {
        const float4* src = (const float4*)(enc + ((size_t)(b * SS + s0)) * DD);
        float4* dst = (float4*)rawA;
        #pragma unroll
        for (int i = 0; i < 16; ++i) dst[tid + i * 256] = src[tid + i * 256];
    }
    for (int i = tid; i < UU; i += 256) { sG[i] = g[b * UU + i]; sV[i] = V[i]; }
    if (tid < 16) sScore[tid] = 0.f;
    __syncthreads();

    // Phase 2: swizzle into WMMA 16-bit A-fragment order (contiguous k, same order as W1p):
    // lane -> M = lane%16, half h = lane/16;
    // e -> K = h*16 + e
    for (int idx = tid; idx < KT * 32 * 16; idx += 256) {
        int e  = idx & 15;
        int L  = (idx >> 4) & 31;
        int kt = idx >> 9;
        int m  = L & 15, h = L >> 4;
        int K  = h * 16 + e;
        sA[idx] = f32_to_bf16(rawA[m * DD + kt * 32 + K]);
    }
    __syncthreads();

    // Phase 3: WMMA K-loop, fully unrolled. Per-lane bases computed once;
    // all loop-varying offsets are compile-time immediates.
    //   A frag  : LDS  base + kt*1024 bytes            (ds 16-bit offset)
    //   B frag  : W1p  base + (j*KT + kt)*1024 bytes   (24-bit IOFFSET)
    const char* aB = (const char*)&sA[lane * 16];
    const char* wB = (const char*)(W1p + ((size_t)(wave * 8 * KT + 0) * 32 + lane) * 16);

    v8f acc[8] = {};
    #pragma unroll
    for (int kt = 0; kt < KT; ++kt) {
        v16bf a = *(const v16bf*)(aB + kt * 1024);
        #pragma unroll
        for (int j = 0; j < 8; ++j) {
            v16bf bb = *(const v16bf*)(wB + (j * KT + kt) * 1024);
            acc[j] = __builtin_amdgcn_wmma_f32_16x16x32_bf16(
                false, a, false, bb, (short)0, acc[j], false, false);
        }
    }

    // Phase 4: fused epilogue. C layout: lane -> N = lane%16,
    // VGPR r -> M = r + 8*(lane/16).  score_row += V[u]*tanh(acc + g[u]).
    float rsum[8] = {};
    #pragma unroll
    for (int j = 0; j < 8; ++j) {
        const int u  = (wave * 8 + j) * 16 + (lane & 15);
        const float gv = sG[u];
        const float vv = sV[u];
        #pragma unroll
        for (int r = 0; r < 8; ++r)
            rsum[r] = fmaf(vv, tanhf(acc[j][r] + gv), rsum[r]);
    }
    // Butterfly across the 16 lanes of each half (masks 1..8 stay in-half).
    #pragma unroll
    for (int m = 8; m >= 1; m >>= 1) {
        #pragma unroll
        for (int r = 0; r < 8; ++r)
            rsum[r] += __shfl_xor(rsum[r], m, 32);
    }
    if ((lane & 15) == 0) {
        const int base = (lane >> 4) * 8;     // lanes 0/16 hold rows 0-7 / 8-15
        #pragma unroll
        for (int r = 0; r < 8; ++r) atomicAdd(&sScore[base + r], rsum[r]);
    }
    __syncthreads();
    if (tid < 16) sc[b * SS + s0 + tid] = sScore[tid] + bvp[0];
}

// ---------------------------------------------------------------------------
// Kernel 3: softmax over S per batch. One block per batch.
// ---------------------------------------------------------------------------
__global__ __launch_bounds__(256) void k_softmax(const float* __restrict__ sc,
                                                 float* __restrict__ w) {
    __shared__ float red[256];
    const int b = blockIdx.x, tid = threadIdx.x;
    float vals[8];
    float mx = -3.402823466e+38f;
    #pragma unroll
    for (int i = 0; i < 8; ++i) {
        vals[i] = sc[b * SS + tid + i * 256];
        mx = fmaxf(mx, vals[i]);
    }
    red[tid] = mx; __syncthreads();
    for (int off = 128; off > 0; off >>= 1) {
        if (tid < off) red[tid] = fmaxf(red[tid], red[tid + off]);
        __syncthreads();
    }
    mx = red[0]; __syncthreads();

    float sum = 0.f;
    #pragma unroll
    for (int i = 0; i < 8; ++i) { vals[i] = __expf(vals[i] - mx); sum += vals[i]; }
    red[tid] = sum; __syncthreads();
    for (int off = 128; off > 0; off >>= 1) {
        if (tid < off) red[tid] += red[tid + off];
        __syncthreads();
    }
    const float inv = 1.0f / red[0];
    #pragma unroll
    for (int i = 0; i < 8; ++i) w[b * SS + tid + i * 256] = vals[i] * inv;
}

// ---------------------------------------------------------------------------
// Kernel 4: ctx[b,d] = sum_s w[b,s] * enc[b,s,d]  (second enc pass, coalesced)
// ---------------------------------------------------------------------------
__global__ __launch_bounds__(256) void k_ctx(const float* __restrict__ enc,
                                             const float* __restrict__ w,
                                             float* __restrict__ out) {
    __shared__ float sw[SS];
    const int b = blockIdx.y, tid = threadIdx.x;
    for (int i = tid; i < SS; i += 256) sw[i] = w[b * SS + i];
    __syncthreads();
    const int d = blockIdx.x * 256 + tid;
    const float* ep = enc + (size_t)b * SS * DD + d;
    float acc = 0.f;
    #pragma unroll 8
    for (int s = 0; s < SS; ++s) acc = fmaf(sw[s], ep[(size_t)s * DD], acc);
    out[b * DD + d] = acc;
}

// ---------------------------------------------------------------------------
// Launch. Workspace layout (needs ~2.7 MB):
//   [0, 2MB)        W1p   bf16 swizzled W1
//   [+, 128KB)      g     fp32 [B,U]
//   [+, 256KB)      sc    fp32 [B,S]
//   [+, 256KB)      w     fp32 [B,S]
// ---------------------------------------------------------------------------
extern "C" void kernel_launch(void* const* d_in, const int* in_sizes, int n_in,
                              void* d_out, int out_size, void* d_ws, size_t ws_size,
                              hipStream_t stream) {
    const float* enc = (const float*)d_in[0];
    const float* hid = (const float*)d_in[1];
    const float* W1  = (const float*)d_in[2];
    const float* b1  = (const float*)d_in[3];
    const float* W2  = (const float*)d_in[4];
    const float* b2  = (const float*)d_in[5];
    const float* V   = (const float*)d_in[6];
    const float* bv  = (const float*)d_in[7];
    float* out = (float*)d_out;

    char* ws = (char*)d_ws;
    unsigned short* W1p = (unsigned short*)ws;                 ws += (size_t)DD * UU * 2;
    float* g  = (float*)ws;                                    ws += (size_t)BB * UU * 4;
    float* sc = (float*)ws;                                    ws += (size_t)BB * SS * 4;
    float* w  = (float*)ws;

    k_w1pack <<<dim3((DD * UU) / 256), 256, 0, stream>>>(W1, W1p);
    k_g      <<<dim3(UU / 256, BB),    256, 0, stream>>>(hid, W2, b1, b2, g);
    k_score  <<<dim3(SS / 16, BB),     256, 0, stream>>>(enc, W1p, g, V, bv, sc);
    k_softmax<<<dim3(BB),              256, 0, stream>>>(sc, w);
    k_ctx    <<<dim3(DD / 256, BB),    256, 0, stream>>>(enc, w, out);
}
